// AttnDecoderRNN_3393024164093
// MI455X (gfx1250) — matrix operand order repaired and checked by reading, not compiled, not run on hardware
//
#include <hip/hip_runtime.h>
#include <hip/hip_bf16.h>
#include <math.h>

// Shapes
#define SEQ   512
#define BATCH 64
#define HD    1024          // H = I = EH
#define HC    2048          // H + EH
#define VOC   50257

typedef __bf16 v16bf __attribute__((ext_vector_type(16)));
typedef float  v8f   __attribute__((ext_vector_type(8)));

#define WMMA_BF16(A, B, C) \
  __builtin_amdgcn_wmma_f32_16x16x32_bf16(false, (A), false, (B), (short)0, (C), false, false)

// ---- fragment loaders (f32 -> bf16 on the fly) -------------------------------
// B-matrix: 16 contiguous K values of one row (lane = N column)
__device__ __forceinline__ v16bf ldB16(const float* __restrict__ p, float scale) {
  v16bf r;
#pragma unroll
  for (int i = 0; i < 16; i += 4) {
    float4 f = *reinterpret_cast<const float4*>(p + i);
    r[i + 0] = (__bf16)(f.x * scale);
    r[i + 1] = (__bf16)(f.y * scale);
    r[i + 2] = (__bf16)(f.z * scale);
    r[i + 3] = (__bf16)(f.w * scale);
  }
  return r;
}
// A-matrix: two 8-element chunks per lane (callers use contiguous k = 16h + s, same order as B)
__device__ __forceinline__ v16bf ldA16(const float* __restrict__ p, int c0, int c1) {
  v16bf r;
  float4 a = *reinterpret_cast<const float4*>(p + c0);
  float4 b = *reinterpret_cast<const float4*>(p + c0 + 4);
  float4 c = *reinterpret_cast<const float4*>(p + c1);
  float4 d = *reinterpret_cast<const float4*>(p + c1 + 4);
  r[0]=(__bf16)a.x; r[1]=(__bf16)a.y; r[2]=(__bf16)a.z;  r[3]=(__bf16)a.w;
  r[4]=(__bf16)b.x; r[5]=(__bf16)b.y; r[6]=(__bf16)b.z;  r[7]=(__bf16)b.w;
  r[8]=(__bf16)c.x; r[9]=(__bf16)c.y; r[10]=(__bf16)c.z; r[11]=(__bf16)c.w;
  r[12]=(__bf16)d.x; r[13]=(__bf16)d.y; r[14]=(__bf16)d.z; r[15]=(__bf16)d.w;
  return r;
}

// ============================ Kernel A: GRU step ==============================
// gx = x @ W_ih^T + b_ih ; h0==0 => gh = b_hh ; h = (1-z)*n
// grid = 64 column-tiles of H, block = 128 (4 waves, one 16-row M-tile each)
__global__ void __launch_bounds__(128)
gru_kernel(const float* __restrict__ x, const float* __restrict__ W_ih,
           const float* __restrict__ b_ih, const float* __restrict__ b_hh,
           float* __restrict__ h_ws, float* __restrict__ out_h) {
  const int tid = threadIdx.x, lane = tid & 31, wave = tid >> 5;
  const int hi = lane >> 4, l15 = lane & 15;
  const int j0 = blockIdx.x * 16;   // hidden-column tile
  const int m0 = wave * 16;         // batch-row tile

  v8f ar, az, an;
#pragma unroll
  for (int i = 0; i < 8; i++) { ar[i] = 0.f; az[i] = 0.f; an[i] = 0.f; }

  const float* xrow = x + (size_t)(m0 + l15) * HD;
  const int rowR = j0 + l15;                       // W_ih row for this lane's N
  const float* wr = W_ih + (size_t)rowR * HD;
  const float* wz = W_ih + (size_t)(HD + rowR) * HD;
  const float* wn = W_ih + (size_t)(2 * HD + rowR) * HD;

  for (int kb = 0; kb < HD; kb += 32) {
    const int c0 = kb + (hi ? 16 : 0);
    const int c1 = kb + 8 + (hi ? 16 : 0);
    v16bf A = ldA16(xrow, c0, c1);
    const int bk = kb + (hi ? 16 : 0);
    v16bf Br = ldB16(wr + bk, 1.f);
    v16bf Bz = ldB16(wz + bk, 1.f);
    v16bf Bn = ldB16(wn + bk, 1.f);
    ar = WMMA_BF16(A, Br, ar);
    az = WMMA_BF16(A, Bz, az);
    an = WMMA_BF16(A, Bn, an);
  }

  const int jc = j0 + l15;
  const float bir = b_ih[jc], biz = b_ih[HD + jc], bin_ = b_ih[2 * HD + jc];
  const float bhr = b_hh[jc], bhz = b_hh[HD + jc], bhn = b_hh[2 * HD + jc];
#pragma unroll
  for (int i = 0; i < 8; i++) {
    const int b = m0 + i + (hi ? 8 : 0);
    float r = 1.f / (1.f + __expf(-(ar[i] + bir + bhr)));
    float z = 1.f / (1.f + __expf(-(az[i] + biz + bhz)));
    float n = tanhf(an[i] + bin_ + r * bhn);
    float hv = (1.f - z) * n;            // + z*h0, h0 == 0
    h_ws[(size_t)b * HD + jc] = hv;
    out_h[(size_t)b * HD + jc] = hv;
  }
}

// =================== Kernel B1: w2[b,c] = sum_j h[b,j]*attn_W[j,c] ============
// grid = (2048/256 c-tiles, 8 b-groups), block = 256; 8 batches per block
__global__ void __launch_bounds__(256)
w2_kernel(const float* __restrict__ h, const float* __restrict__ attn_W,
          float* __restrict__ w2) {
  __shared__ float hs[8][HD];
  const int tid = threadIdx.x;
  const int c = blockIdx.x * 256 + tid;
  const int b0 = blockIdx.y * 8;
#pragma unroll
  for (int bb = 0; bb < 8; bb++)
    for (int i = tid; i < HD; i += 256) hs[bb][i] = h[(size_t)(b0 + bb) * HD + i];
  __syncthreads();
  float acc[8];
#pragma unroll
  for (int bb = 0; bb < 8; bb++) acc[bb] = 0.f;
  for (int j = 0; j < HD; j++) {
    const float wv = attn_W[(size_t)j * HC + c];
#pragma unroll
    for (int bb = 0; bb < 8; bb++) acc[bb] += hs[bb][j] * wv;
  }
#pragma unroll
  for (int bb = 0; bb < 8; bb++) w2[(size_t)(b0 + bb) * HC + c] = acc[bb];
}

// ============ Kernel B2: c_b = sum_h h[b,h]*(w2[b,h] + attn_b[h]) =============
__global__ void __launch_bounds__(256)
cb_kernel(const float* __restrict__ h, const float* __restrict__ w2,
          const float* __restrict__ attn_b, float* __restrict__ cvals) {
  __shared__ float red[256];
  const int b = blockIdx.x, tid = threadIdx.x;
  float acc = 0.f;
  for (int i = tid; i < HD; i += 256)
    acc += h[(size_t)b * HD + i] * (w2[(size_t)b * HC + i] + attn_b[i]);
  red[tid] = acc; __syncthreads();
  for (int o = 128; o > 0; o >>= 1) { if (tid < o) red[tid] += red[tid + o]; __syncthreads(); }
  if (tid == 0) cvals[b] = red[0];
}

// ======== Kernel C: scores[b,s] = c_b + enc[s,b]·v[b] ; softmax over s ========
// grid = 64 (one block per batch), block = 256 (8 waves)
__global__ void __launch_bounds__(256)
attn_kernel(const float* __restrict__ enc, const float* __restrict__ w2,
            const float* __restrict__ cvals, float* __restrict__ aw_ws,
            float* __restrict__ out_aw) {
  __shared__ float vsh[HD];
  __shared__ float sc[SEQ];
  __shared__ float red[256];
  const int b = blockIdx.x, tid = threadIdx.x, lane = tid & 31, wave = tid >> 5;
  for (int i = tid; i < HD; i += 256) vsh[i] = w2[(size_t)b * HC + HD + i];
  __syncthreads();
  const float cb = cvals[b];
  for (int s = wave; s < SEQ; s += 8) {
    const float* erow = enc + ((size_t)s * BATCH + b) * HD;
    float acc = 0.f;
#pragma unroll 8
    for (int i = 0; i < 32; i++) { const int e = i * 32 + lane; acc += erow[e] * vsh[e]; }
    for (int off = 16; off > 0; off >>= 1) acc += __shfl_xor(acc, off, 32);
    if (lane == 0) sc[s] = cb + acc;
  }
  __syncthreads();
  float m = -INFINITY;
  for (int s = tid; s < SEQ; s += 256) m = fmaxf(m, sc[s]);
  red[tid] = m; __syncthreads();
  for (int o = 128; o > 0; o >>= 1) { if (tid < o) red[tid] = fmaxf(red[tid], red[tid + o]); __syncthreads(); }
  m = red[0]; __syncthreads();
  float su = 0.f;
  for (int s = tid; s < SEQ; s += 256) su += __expf(sc[s] - m);
  red[tid] = su; __syncthreads();
  for (int o = 128; o > 0; o >>= 1) { if (tid < o) red[tid] += red[tid + o]; __syncthreads(); }
  const float inv = 1.f / red[0];
  for (int s = tid; s < SEQ; s += 256) {
    const float w = __expf(sc[s] - m) * inv;
    aw_ws[(size_t)b * SEQ + s] = w;
    out_aw[(size_t)b * SEQ + s] = w;
  }
}

// ============= Kernel D: context[b,e] = sum_s w[b,s] * enc[s,b,e] =============
// grid = (HD/256, 64), block = 256
__global__ void __launch_bounds__(256)
ctx_kernel(const float* __restrict__ enc, const float* __restrict__ aw,
           float* __restrict__ ctx) {
  __shared__ float lw[SEQ];
  const int tid = threadIdx.x, b = blockIdx.y;
  const int e = blockIdx.x * 256 + tid;
  for (int s = tid; s < SEQ; s += 256) lw[s] = aw[(size_t)b * SEQ + s];
  __syncthreads();
  float acc = 0.f;
  for (int s = 0; s < SEQ; s++) acc += lw[s] * enc[((size_t)s * BATCH + b) * HD + e];
  ctx[(size_t)b * HD + e] = acc;
}

// ====== Kernel E: dec = [h|ctx] @ out_W^T + out_b  (WMMA bf16, raw logits) ====
// grid = ceil(V/16) = 3142 N-tiles, block = 128 (4 waves, one M-tile each)
__global__ void __launch_bounds__(128)
outgemm_kernel(const float* __restrict__ h, const float* __restrict__ ctx,
               const float* __restrict__ out_W, const float* __restrict__ out_b,
               float* __restrict__ dec) {
  const int tid = threadIdx.x, lane = tid & 31, wave = tid >> 5;
  const int hi = lane >> 4, l15 = lane & 15;
  const int n0 = blockIdx.x * 16;
  const int m0 = wave * 16;
  const int nrow = n0 + l15;
  const int nclamp = nrow < VOC ? nrow : (VOC - 1);
  const float bscale = nrow < VOC ? 1.f : 0.f;   // zero pad columns, EXEC stays full
  const float* wrow = out_W + (size_t)nclamp * HC;
  const size_t arow_off = (size_t)(m0 + l15) * HD;

  v8f acc;
#pragma unroll
  for (int i = 0; i < 8; i++) acc[i] = 0.f;

  for (int kb = 0; kb < HC; kb += 32) {
    const float* arow = (kb < HD) ? (h + arow_off + kb) : (ctx + arow_off + (kb - HD));
    const int c0 = (hi ? 16 : 0);
    const int c1 = 8 + (hi ? 16 : 0);
    v16bf A = ldA16(arow, c0, c1);
    v16bf B = ldB16(wrow + kb + (hi ? 16 : 0), bscale);
    acc = WMMA_BF16(A, B, acc);
  }

  if (nrow < VOC) {
    const float ob = out_b[nrow];
#pragma unroll
    for (int i = 0; i < 8; i++) {
      const int b = m0 + i + (hi ? 8 : 0);
      dec[(size_t)b * VOC + nrow] = acc[i] + ob;
    }
  }
}

// =================== Kernel F: logZ[b] = logsumexp(dec[b,:]) ==================
__global__ void __launch_bounds__(256)
logz_kernel(const float* __restrict__ dec, float* __restrict__ logz) {
  __shared__ float red[256];
  const int b = blockIdx.x, tid = threadIdx.x;
  const float* row = dec + (size_t)b * VOC;
  float m = -INFINITY;
  for (int i = tid; i < VOC; i += 256) m = fmaxf(m, row[i]);
  red[tid] = m; __syncthreads();
  for (int o = 128; o > 0; o >>= 1) { if (tid < o) red[tid] = fmaxf(red[tid], red[tid + o]); __syncthreads(); }
  m = red[0]; __syncthreads();
  float s = 0.f;
  for (int i = tid; i < VOC; i += 256) s += expf(row[i] - m);
  red[tid] = s; __syncthreads();
  for (int o = 128; o > 0; o >>= 1) { if (tid < o) red[tid] += red[tid + o]; __syncthreads(); }
  if (tid == 0) logz[b] = m + logf(red[0]);
}

// =================== Kernel G: out[b,n] -= logZ[b] (in place) =================
__global__ void __launch_bounds__(256)
sub_kernel(float* __restrict__ dec, const float* __restrict__ logz) {
  const size_t idx = (size_t)blockIdx.x * 256 + threadIdx.x;
  if (idx < (size_t)BATCH * VOC) dec[idx] -= logz[idx / VOC];
}

// ==============================================================================
extern "C" void kernel_launch(void* const* d_in, const int* in_sizes, int n_in,
                              void* d_out, int out_size, void* d_ws, size_t ws_size,
                              hipStream_t stream) {
  const float* x      = (const float*)d_in[0];   // (1,64,1024)
  const float* enc    = (const float*)d_in[1];   // (512,64,1024)
  const float* W_ih   = (const float*)d_in[3];   // (3072,1024)
  const float* b_ih   = (const float*)d_in[5];
  const float* b_hh   = (const float*)d_in[6];
  const float* attn_W = (const float*)d_in[7];   // (1024,2048)
  const float* attn_b = (const float*)d_in[8];
  const float* out_W  = (const float*)d_in[9];   // (50257,2048)
  const float* out_b  = (const float*)d_in[10];

  float* out  = (float*)d_out;
  float* dec  = out;                                     // (64,50257)
  float* outh = out + (size_t)BATCH * VOC;               // (1,64,1024)
  float* outa = outh + (size_t)BATCH * HD;               // (64,512)

  float* W   = (float*)d_ws;
  float* h   = W;                        // 65536
  float* w2  = h + BATCH * HD;           // 131072
  float* cv  = w2 + BATCH * HC;          // 64
  float* aw  = cv + BATCH;               // 32768
  float* ctx = aw + BATCH * SEQ;         // 65536
  float* lz  = ctx + BATCH * HD;         // 64

  gru_kernel<<<64, 128, 0, stream>>>(x, W_ih, b_ih, b_hh, h, outh);
  w2_kernel<<<dim3(HC / 256, BATCH / 8), 256, 0, stream>>>(h, attn_W, w2);
  cb_kernel<<<BATCH, 256, 0, stream>>>(h, w2, attn_b, cv);
  attn_kernel<<<BATCH, 256, 0, stream>>>(enc, w2, cv, aw, outa);
  ctx_kernel<<<dim3(HD / 256, BATCH), 256, 0, stream>>>(enc, aw, ctx);
  outgemm_kernel<<<(VOC + 15) / 16, 128, 0, stream>>>(h, ctx, out_W, out_b, dec);
  logz_kernel<<<BATCH, 256, 0, stream>>>(dec, lz);
  sub_kernel<<<((size_t)BATCH * VOC + 255) / 256, 256, 0, stream>>>(dec, lz);
}
